// CrossAttention_24876450578750
// MI455X (gfx1250) — matrix operand order repaired and checked by reading, not compiled, not run on hardware
//
#include <hip/hip_runtime.h>
#include <hip/hip_bf16.h>

// bf16 / fp32 vectors matching CDNA5 WMMA operand shapes
typedef __attribute__((ext_vector_type(16))) __bf16 v16bf;
typedef __attribute__((ext_vector_type(8)))  float  v8f;

constexpr int   kB     = 16;
constexpr int   kS     = 2048;
constexpr int   kD     = 1024;
constexpr float kScale = 0.03125f;   // 1/sqrt(1024)
constexpr float kEps   = 1e-6f;

constexpr int kSsStride = 72;        // padded row stride (floats) for score tile
constexpr int kPsStride = 80;        // padded row stride (bf16) for P tile

union BFrag { uint4  q[2]; v16bf f; };   // 32B fragment = two b128 loads

__device__ __forceinline__ v8f wmma_bf16(v16bf a, v16bf b, v8f c) {
  return __builtin_amdgcn_wmma_f32_16x16x32_bf16(
      /*neg_a=*/false, a, /*neg_b=*/false, b,
      /*c_mod=*/(short)0, c, /*reuse_a=*/false, /*reuse_b=*/false);
}

// Build a 16-element bf16 fragment from two 16B-aligned 16-byte runs.
__device__ __forceinline__ v16bf load_frag(const __bf16* p0, const __bf16* p1) {
  BFrag fu;
  fu.q[0] = *reinterpret_cast<const uint4*>(p0);
  fu.q[1] = *reinterpret_cast<const uint4*>(p1);
  return fu.f;
}

// ---------------------------------------------------------------------------
// Kernel 0: WT[n][k] = bf16(W[k][n])  (LDS-tiled transpose + convert, 32x32)
// ---------------------------------------------------------------------------
__global__ void __launch_bounds__(256)
transpose_w_bf16(const float* __restrict__ W, __bf16* __restrict__ WT) {
  __shared__ float t[32][33];
  const int tx = threadIdx.x & 31;
  const int ty = threadIdx.x >> 5;            // 0..7
  const int k0 = blockIdx.x * 32;
  const int n0 = blockIdx.y * 32;
#pragma unroll
  for (int j = 0; j < 4; ++j)
    t[ty + 8 * j][tx] = W[(size_t)(k0 + ty + 8 * j) * kD + n0 + tx];
  __syncthreads();
#pragma unroll
  for (int j = 0; j < 4; ++j)
    WT[(size_t)(n0 + ty + 8 * j) * kD + k0 + tx] = (__bf16)t[tx][ty + 8 * j];
}

// ---------------------------------------------------------------------------
// Kernel 1: Out = bf16( X @ W + bias ), W pre-transposed in bf16 (WT[n][k]).
// Block = 128 thr (4 waves), block tile 16(M) x 256(N), wave tile 16x64.
// TRANSPOSE!=0 stores Out as [B][D][S] (used for V).  A-fragment pipelined
// across k-steps; B-fragments double-buffered across n-tiles.
// ---------------------------------------------------------------------------
template <int TRANSPOSE>
__global__ void __launch_bounds__(128)
proj_gemm_bf16(const float* __restrict__ X, const __bf16* __restrict__ WT,
               const float* __restrict__ bias, __bf16* __restrict__ Out) {
  const int lane = threadIdx.x & 31;
  const int wave = threadIdx.x >> 5;
  const int hf   = lane >> 4;
  const int li   = lane & 15;
  const int row0 = blockIdx.x * 16;
  const int col0 = blockIdx.y * 256 + wave * 64;

  v8f acc[4] = {};
  const float* xrow = X + (size_t)(row0 + li) * kD;

  auto loadA = [&](int k0) {
    BFrag a;
#pragma unroll
    for (int g = 0; g < 2; ++g) {
      const float* rp = xrow + k0 + hf * 16 + g * 8;
      float4 f0 = *reinterpret_cast<const float4*>(rp);
      float4 f1 = *reinterpret_cast<const float4*>(rp + 4);
      a.f[g * 8 + 0] = (__bf16)f0.x;  a.f[g * 8 + 1] = (__bf16)f0.y;
      a.f[g * 8 + 2] = (__bf16)f0.z;  a.f[g * 8 + 3] = (__bf16)f0.w;
      a.f[g * 8 + 4] = (__bf16)f1.x;  a.f[g * 8 + 5] = (__bf16)f1.y;
      a.f[g * 8 + 6] = (__bf16)f1.z;  a.f[g * 8 + 7] = (__bf16)f1.w;
    }
    return a.f;
  };

  v16bf a_cur = loadA(0);
  for (int k0 = 0; k0 < kD; k0 += 32) {
    const int knext = (k0 + 32 < kD) ? (k0 + 32) : k0;   // clamped (safe dup)
    const __bf16* wp0 = WT + (size_t)(col0 + li) * kD + k0 + hf * 16;
    v16bf b_cur = load_frag(wp0, wp0 + 8);
    v16bf a_nxt = loadA(knext);
#pragma unroll
    for (int nt = 0; nt < 4; ++nt) {
      v16bf b_nxt = {};
      if (nt < 3) {
        const __bf16* wp = WT + (size_t)(col0 + (nt + 1) * 16 + li) * kD + k0 + hf * 16;
        b_nxt = load_frag(wp, wp + 8);
      }
      acc[nt] = wmma_bf16(a_cur, b_cur, acc[nt]);
      b_cur = b_nxt;
    }
    a_cur = a_nxt;
  }

#pragma unroll
  for (int nt = 0; nt < 4; ++nt) {
    int   col = col0 + nt * 16 + li;
    float bb  = bias[col];
#pragma unroll
    for (int r = 0; r < 8; ++r) {
      int    rowg = row0 + r + 8 * hf;           // global row in [0, B*S)
      __bf16 val  = (__bf16)(acc[nt][r] + bb);
      if (TRANSPOSE) {
        int bb2 = rowg >> 11;                    // rowg / kS
        int s   = rowg & (kS - 1);
        Out[((size_t)bb2 * kD + col) * kS + s] = val;
      } else {
        Out[(size_t)rowg * kD + col] = val;
      }
    }
  }
}

// ---------------------------------------------------------------------------
// Kernel 2: flash attention.  Block = 128 thr (4 waves) per (batch, 16-row
// query tile).  Q tile staged once into LDS via global_load_async_to_lds_b128
// (ASYNCcnt).  Per 64-key tile:
//   - wave w computes full-D scores for kv cols [w*16,w*16+16) (32 WMMA,
//     software-pipelined fragment loads)
//   - all 4 waves cooperate on the online softmax (4 rows/wave, shfl_xor
//     butterflies for row max/sum), P written as packed b128
//   - every wave rescales its O slice (256 cols = 16 C tiles) and runs
//     O += P @ V^T-slice (32 WMMA, double-buffered b128 V fragments)
// Epilogue: out = O/(l+eps) + x.
// ---------------------------------------------------------------------------
__global__ void __launch_bounds__(128)
attn_flash(const __bf16* __restrict__ Q, const __bf16* __restrict__ Km,
           const __bf16* __restrict__ Vt, const float* __restrict__ x,
           float* __restrict__ out) {
  __shared__ __bf16 Qs[16 * kD];             // query tile, [row][d]
  __shared__ float  Ss[16 * kSsStride];      // score tile, [row][col] (padded)
  __shared__ __bf16 Ps[16 * kPsStride];      // probabilities, [row][k] (padded)
  __shared__ float  CorrS[16];
  __shared__ float  LSum[16];

  const int tid  = threadIdx.x;
  const int lane = tid & 31;
  const int wave = tid >> 5;                 // 0..3
  const int hf   = lane >> 4;
  const int li   = lane & 15;
  const int b    = blockIdx.y;
  const int q0   = blockIdx.x * 16;

  // ---- stage Q tile (16 x 1024 bf16 = 2048 x b128) via async loads ----
  {
    const uint4* src = reinterpret_cast<const uint4*>(Q + ((size_t)b * kS + q0) * kD);
    unsigned     ldsbase = (unsigned)(uintptr_t)(&Qs[0]);
#pragma unroll
    for (int i = 0; i < 16; ++i) {
      int          f = tid + i * 128;
      unsigned     lds = ldsbase + f * 16;
      const uint4* g   = src + f;
      asm volatile("global_load_async_to_lds_b128 %0, %1, off"
                   :: "v"(lds), "v"(g) : "memory");
    }
    asm volatile("s_wait_asynccnt 0x0" ::: "memory");
  }
  __syncthreads();

  v8f o[16];
#pragma unroll
  for (int nt = 0; nt < 16; ++nt) o[nt] = (v8f){};
  float m_run = -1e30f, l_run = 0.f;   // per-lane state for its softmax row

  const int srow = wave * 4 + (lane & 3);    // softmax row owned by this lane
  const int c0   = (lane >> 2) * 8;          // 8 softmax columns per lane

  for (int kv0 = 0; kv0 < kS; kv0 += 64) {
    const int kvb = kv0 + wave * 16;

    // prefetch next tile's K row / V column for this lane (global_prefetch_b8)
    if (kv0 + 64 < kS) {
      __builtin_prefetch(Km + ((size_t)b * kS + kvb + 64 + li) * kD + hf * 16, 0, 1);
      __builtin_prefetch(Vt + ((size_t)b * kD + wave * 256 + li) * kS + kv0 + 64 + hf * 16, 0, 1);
    }

    // ---- scores: s(16x16) = Q(16x1024) @ K_slice^T, pipelined ----
    v8f s = {};
    const __bf16* krow = Km + ((size_t)b * kS + kvb + li) * kD;
    const __bf16* qp0  = &Qs[li * kD + hf * 8];
    v16bf a_cur = load_frag(qp0, qp0 + 16);
    v16bf k_cur = load_frag(krow + hf * 8, krow + hf * 8 + 16);
#pragma unroll 4
    for (int ks = 0; ks < 31; ++ks) {
      const __bf16* qp = &Qs[li * kD + (ks + 1) * 32 + hf * 8];
      v16bf a_nxt = load_frag(qp, qp + 16);
      const __bf16* kp = krow + (ks + 1) * 32 + hf * 8;
      v16bf k_nxt = load_frag(kp, kp + 16);
      s = wmma_bf16(a_cur, k_cur, s);
      a_cur = a_nxt;
      k_cur = k_nxt;
    }
    s = wmma_bf16(a_cur, k_cur, s);
#pragma unroll
    for (int r = 0; r < 8; ++r)
      Ss[(8 * hf + r) * kSsStride + wave * 16 + li] = s[r];
    __syncthreads();

    // ---- online softmax: 4 rows per wave, 8 lanes x 8 cols per row ----
    {
      float4 f0 = *reinterpret_cast<const float4*>(&Ss[srow * kSsStride + c0]);
      float4 f1 = *reinterpret_cast<const float4*>(&Ss[srow * kSsStride + c0 + 4]);
      float vals[8] = {f0.x * kScale, f0.y * kScale, f0.z * kScale, f0.w * kScale,
                       f1.x * kScale, f1.y * kScale, f1.z * kScale, f1.w * kScale};
      float mt = vals[0];
#pragma unroll
      for (int j = 1; j < 8; ++j) mt = fmaxf(mt, vals[j]);
      mt = fmaxf(mt, __shfl_xor(mt, 4, 32));
      mt = fmaxf(mt, __shfl_xor(mt, 8, 32));
      mt = fmaxf(mt, __shfl_xor(mt, 16, 32));
      float m_new = fmaxf(m_run, mt);
      float corr  = __expf(m_run - m_new);
      float ls    = 0.f;
      union { uint4 q; __bf16 h[8]; } pk;
#pragma unroll
      for (int j = 0; j < 8; ++j) {
        float p = __expf(vals[j] - m_new);
        ls += p;
        pk.h[j] = (__bf16)p;
      }
      *reinterpret_cast<uint4*>(&Ps[srow * kPsStride + c0]) = pk.q;
      ls += __shfl_xor(ls, 4, 32);
      ls += __shfl_xor(ls, 8, 32);
      ls += __shfl_xor(ls, 16, 32);
      l_run = l_run * corr + ls;
      m_run = m_new;
      if ((lane >> 2) == 0) CorrS[srow] = corr;
    }
    __syncthreads();

    // ---- rescale O, then O += P @ V (double-buffered V fragments) ----
    float corrv = (lane < 16) ? CorrS[lane] : 0.f;
#pragma unroll
    for (int r = 0; r < 8; ++r) {
      float cf = __shfl(corrv, r + 8 * hf, 32);
#pragma unroll
      for (int nt = 0; nt < 16; ++nt) o[nt][r] *= cf;
    }
    const __bf16* vbase = Vt + (size_t)b * kD * kS + ((size_t)(wave * 256 + li)) * kS + kv0;
#pragma unroll
    for (int ks = 0; ks < 2; ++ks) {
      const __bf16* pp = &Ps[li * kPsStride + ks * 32 + hf * 16];
      v16bf pa = load_frag(pp, pp + 8);
      const __bf16* vp0 = vbase + ks * 32 + hf * 16;
      v16bf vb_cur = load_frag(vp0, vp0 + 8);
#pragma unroll
      for (int nt = 0; nt < 16; ++nt) {
        v16bf vb_nxt = {};
        if (nt < 15) {
          const __bf16* vp = vp0 + (size_t)(nt + 1) * 16 * kS;
          vb_nxt = load_frag(vp, vp + 8);
        }
        o[nt] = wmma_bf16(pa, vb_cur, o[nt]);
        vb_cur = vb_nxt;
      }
    }
  }

  // ---- epilogue: divide by (l+eps), add residual ----
  if ((lane >> 2) == 0) LSum[srow] = l_run;
  __syncthreads();
  float lv = (lane < 16) ? LSum[lane] : 0.f;
#pragma unroll
  for (int r = 0; r < 8; ++r) {
    float inv = 1.f / (__shfl(lv, r + 8 * hf, 32) + kEps);
    int   row = q0 + r + 8 * hf;
#pragma unroll
    for (int nt = 0; nt < 16; ++nt) {
      int    col = wave * 256 + nt * 16 + li;
      size_t idx = ((size_t)b * kS + row) * kD + col;
      out[idx]   = o[nt][r] * inv + x[idx];
    }
  }
}

// ---------------------------------------------------------------------------
extern "C" void kernel_launch(void* const* d_in, const int* in_sizes, int n_in,
                              void* d_out, int out_size, void* d_ws, size_t ws_size,
                              hipStream_t stream) {
  const float* x  = (const float*)d_in[0];
  const float* y  = (const float*)d_in[1];
  const float* Wq = (const float*)d_in[2];
  const float* bq = (const float*)d_in[3];
  const float* Wk = (const float*)d_in[4];
  const float* bk = (const float*)d_in[5];
  const float* Wv = (const float*)d_in[6];
  const float* bv = (const float*)d_in[7];
  float* out = (float*)d_out;

  const size_t elems = (size_t)kB * kS * kD;
  const size_t welem = (size_t)kD * kD;
  __bf16* Qb  = (__bf16*)d_ws;
  __bf16* Kb  = Qb + elems;
  __bf16* Vb  = Kb + elems;          // stored transposed: [B][D][S]
  __bf16* WqT = Vb + elems;          // [n][k] bf16
  __bf16* WkT = WqT + welem;
  __bf16* WvT = WkT + welem;

  dim3 gt(kD / 32, kD / 32, 1);
  transpose_w_bf16<<<gt, 256, 0, stream>>>(Wq, WqT);
  transpose_w_bf16<<<gt, 256, 0, stream>>>(Wk, WkT);
  transpose_w_bf16<<<gt, 256, 0, stream>>>(Wv, WvT);

  dim3 gp(kB * kS / 16, kD / 256, 1);
  proj_gemm_bf16<0><<<gp, 128, 0, stream>>>(x, WqT, bq, Qb);
  proj_gemm_bf16<0><<<gp, 128, 0, stream>>>(y, WkT, bk, Kb);
  proj_gemm_bf16<1><<<gp, 128, 0, stream>>>(y, WvT, bv, Vb);

  dim3 ga(kS / 16, kB, 1);
  attn_flash<<<ga, 128, 0, stream>>>(Qb, Kb, Vb, x, out);
}
